// MultilayerGRU_9113920602658
// MI455X (gfx1250) — matrix operand order repaired and checked by reading, not compiled, not run on hardware
//
#include <hip/hip_runtime.h>
#include <hip/hip_bf16.h>

// ---------------------------------------------------------------------------
// Multilayer GRU on gfx1250 via v_wmma_f32_16x16x32_f16.
// B=64, S=1024, I=128, HID=1024, L=2, O=128.
// Internal row ordering is TIME-MAJOR: m' = t*B + b, so the per-timestep gate
// slices read inside the serial scan are contiguous 128KB blocks.
// ---------------------------------------------------------------------------

typedef _Float16 h16;
typedef __attribute__((ext_vector_type(16))) _Float16 v16h;
typedef __attribute__((ext_vector_type(8)))  _Float16 v8h;
typedef __attribute__((ext_vector_type(8)))  float    v8f;

union Frag16 { v16h v; v8h h[2]; };

#define WMMA_F16(a, b, c) \
  __builtin_amdgcn_wmma_f32_16x16x32_f16(false, (a), false, (b), (short)0, (c), false, false)

// A-fragment: 16x32 f16 tile, row m = m0 + (lane&15).
// ISA layout: lanes 0-15 hold K {k0..k0+7, k0+16..k0+23}; lanes 16-31 add +8.
__device__ __forceinline__ v16h loadA(const h16* __restrict__ A, int lda,
                                      int m0, int k0, int lane) {
  int row  = m0 + (lane & 15);
  int koff = k0 + ((lane >> 4) << 3);
  const h16* p = A + row * lda + koff;
  Frag16 f;
  f.h[0] = *(const v8h*)(p);
  f.h[1] = *(const v8h*)(p + 16);
  return f.v;
}

// B-fragment for C = A * W^T: column n of B == row n of W (row-major, ld=K).
// lanes 0-15: K {k0..k0+7, k0+16..k0+23}; lanes 16-31 add +8 (matches loadA).
__device__ __forceinline__ v16h loadW(const h16* __restrict__ W, int ldw,
                                      int n0, int k0, int lane) {
  int col = n0 + (lane & 15);
  int kb  = k0 + ((lane >> 4) << 3);
  const h16* p = W + col * ldw + kb;
  Frag16 f;
  f.h[0] = *(const v8h*)(p);
  f.h[1] = *(const v8h*)(p + 16);
  return f.v;
}

__device__ __forceinline__ float sigmoidf_(float x) {
  return 1.0f / (1.0f + __expf(-x));
}

// ---------------------------------------------------------------------------
// Generic GEMM: out[m,n] = sum_k A[m,k] * W[n,k] + bias[n]
// A: (M x K) f16 row-major (lda == K).  W: (N x K) f16 row-major.
// One wave computes a 32x64 tile: 2 A-frags, 8 accumulators (8 WMMA / k-chunk).
// Grid must supply exactly (M/32)*(N/64) waves; no divergence (EXEC all 1s).
// PERM_OUT: remap time-major row m'=t*64+b to reference order (b*1024+t).
// ---------------------------------------------------------------------------
template <bool F16OUT, bool PERM_OUT>
__global__ void gemm_bias(const h16* __restrict__ A, const h16* __restrict__ W,
                          const float* __restrict__ bias, void* __restrict__ out,
                          int N, int K) {
  int wid  = (blockIdx.x * blockDim.x + threadIdx.x) >> 5;
  int lane = threadIdx.x & 31;
  int cq = N >> 6;
  int m0 = (wid / cq) << 5;
  int n0 = (wid % cq) << 6;

  v8f acc[2][4] = {};
  for (int k0 = 0; k0 < K; k0 += 32) {
    v16h a0 = loadA(A, K, m0,      k0, lane);
    v16h a1 = loadA(A, K, m0 + 16, k0, lane);
#pragma unroll
    for (int j = 0; j < 4; ++j) {
      v16h b = loadW(W, K, n0 + (j << 4), k0, lane);
      acc[0][j] = WMMA_F16(a0, b, acc[0][j]);
      acc[1][j] = WMMA_F16(a1, b, acc[1][j]);
    }
  }
  // C layout: VGPR v -> row m0 + v + 8*(lane>=16); col n0 + (lane&15)
  int cb = lane & 15;
#pragma unroll
  for (int r = 0; r < 2; ++r) {
    int rbase = m0 + (r << 4) + ((lane >> 4) << 3);
#pragma unroll
    for (int j = 0; j < 4; ++j) {
      int col = n0 + (j << 4) + cb;
      float bv = bias[col];
#pragma unroll
      for (int v = 0; v < 8; ++v) {
        int row = rbase + v;
        // time-major m' = t*64 + b  ->  reference row b*1024 + t
        int orow = PERM_OUT ? (((row & 63) << 10) | (row >> 6)) : row;
        float val = acc[r][j][v] + bv;
        if (F16OUT) ((h16*)out)[orow * N + col] = (h16)val;
        else        ((float*)out)[orow * N + col] = val;
      }
    }
  }
}

// ---------------------------------------------------------------------------
// GRU step phase 1: z = sigmoid(xz_t + h Whz^T), r = sigmoid(xr_t + h Whr^T),
// rh = r*h.   M=64 (batch), N=HID, K=HID.  Split-K x4: block = 4 waves, one
// 16x16 tile, each wave does K-chunk of 256; partial sums reduced via LDS.
// Grid (64, 4) x 128 threads -> 1024 waves per step phase.
// ---------------------------------------------------------------------------
__global__ void gru_step_zr(const h16* __restrict__ hH,
                            const h16* __restrict__ Wz, const h16* __restrict__ Wr,
                            const h16* __restrict__ xz, const h16* __restrict__ xr,
                            const float* __restrict__ hF,
                            float* __restrict__ zF, h16* __restrict__ rh, int t) {
  const int HID = 1024;
  __shared__ float red[3][2][256];
  int lane = threadIdx.x & 31;
  int w    = threadIdx.x >> 5;
  int n0 = blockIdx.x << 4;
  int m0 = blockIdx.y << 4;
  int kbeg = w << 8;

  v8f az = {}, ar = {};
  for (int k0 = kbeg; k0 < kbeg + 256; k0 += 32) {
    v16h a  = loadA(hH, HID, m0, k0, lane);
    v16h bz = loadW(Wz, HID, n0, k0, lane);
    v16h br = loadW(Wr, HID, n0, k0, lane);
    az = WMMA_F16(a, bz, az);
    ar = WMMA_F16(a, br, ar);
  }
  if (w) {
#pragma unroll
    for (int v = 0; v < 8; ++v) {
      red[w - 1][0][(v << 5) | lane] = az[v];
      red[w - 1][1][(v << 5) | lane] = ar[v];
    }
  }
  __syncthreads();
  if (w == 0) {
#pragma unroll
    for (int u = 0; u < 3; ++u)
#pragma unroll
      for (int v = 0; v < 8; ++v) {
        az[v] += red[u][0][(v << 5) | lane];
        ar[v] += red[u][1][(v << 5) | lane];
      }
    int rbase = m0 + ((lane >> 4) << 3);
    int col = n0 + (lane & 15);
#pragma unroll
    for (int v = 0; v < 8; ++v) {
      int b = rbase + v;                           // batch index 0..63
      int gidx = (((t << 6) | b) << 10) | col;     // time-major gate slice
      int hidx = (b << 10) | col;
      float z = sigmoidf_(az[v] + (float)xz[gidx]);
      float r = sigmoidf_(ar[v] + (float)xr[gidx]);
      zF[hidx] = z;
      rh[hidx] = (h16)(r * hF[hidx]);
    }
  }
}

// ---------------------------------------------------------------------------
// GRU step phase 2: g = tanh(xg_t + (r*h) Whg^T); h = z*h + (1-z)*g.
// Same split-K x4 structure.  Writes h (f32 state), h (f16 next-step A
// operand), and the time-major layer output at step t.
// ---------------------------------------------------------------------------
__global__ void gru_step_g(const h16* __restrict__ rh, const h16* __restrict__ Wg,
                           const h16* __restrict__ xg, const float* __restrict__ zF,
                           float* __restrict__ hF, h16* __restrict__ hH,
                           h16* __restrict__ curOut, int t) {
  const int HID = 1024;
  __shared__ float red[3][256];
  int lane = threadIdx.x & 31;
  int w    = threadIdx.x >> 5;
  int n0 = blockIdx.x << 4;
  int m0 = blockIdx.y << 4;
  int kbeg = w << 8;

  v8f ag = {};
  for (int k0 = kbeg; k0 < kbeg + 256; k0 += 32) {
    v16h a = loadA(rh, HID, m0, k0, lane);
    v16h b = loadW(Wg, HID, n0, k0, lane);
    ag = WMMA_F16(a, b, ag);
  }
  if (w) {
#pragma unroll
    for (int v = 0; v < 8; ++v) red[w - 1][(v << 5) | lane] = ag[v];
  }
  __syncthreads();
  if (w == 0) {
#pragma unroll
    for (int u = 0; u < 3; ++u)
#pragma unroll
      for (int v = 0; v < 8; ++v) ag[v] += red[u][(v << 5) | lane];
    int rbase = m0 + ((lane >> 4) << 3);
    int col = n0 + (lane & 15);
#pragma unroll
    for (int v = 0; v < 8; ++v) {
      int b = rbase + v;
      int gidx = (((t << 6) | b) << 10) | col;
      int hidx = (b << 10) | col;
      float g = tanhf(ag[v] + (float)xg[gidx]);
      float z = zF[hidx];
      float hn = z * hF[hidx] + (1.0f - z) * g;
      hF[hidx] = hn;
      hH[hidx] = (h16)hn;
      curOut[gidx] = (h16)hn;
    }
  }
}

// ---------------------------------------------------------------------------
// Utility kernels
// ---------------------------------------------------------------------------
__global__ void cvt_f16(const float* __restrict__ s, h16* __restrict__ d, int n) {
  int i = blockIdx.x * blockDim.x + threadIdx.x;
  if (i < n) d[i] = (h16)s[i];
}

// x (B,S,I) f32 -> f16 with rows permuted to time-major m' = t*64 + b.
__global__ void x_cvt_tm(const float* __restrict__ x, h16* __restrict__ d) {
  int i = blockIdx.x * blockDim.x + threadIdx.x;   // over 64*1024*128
  int col = i & 127;
  int m = i >> 7;            // m' = t*64 + b
  int b = m & 63, t = m >> 6;
  d[i] = (h16)x[(((b << 10) | t) << 7) | col];     // src row b*S + t
}

__global__ void h_init(const float* __restrict__ h0, float* __restrict__ hF,
                       h16* __restrict__ hH, int layer) {
  int i = blockIdx.x * blockDim.x + threadIdx.x;   // 0..65535
  int b = i >> 10, n = i & 1023;
  float v = h0[((b << 1) + layer) << 10 | n];      // h0[b, layer, n], L=2
  hF[i] = v;
  hH[i] = (h16)v;
}

__global__ void h_last(const float* __restrict__ hF, float* __restrict__ outH, int layer) {
  int i = blockIdx.x * blockDim.x + threadIdx.x;
  int b = i >> 10, n = i & 1023;
  outH[((b << 1) + layer) << 10 | n] = hF[i];
}

// ---------------------------------------------------------------------------
// Host launcher
// ---------------------------------------------------------------------------
extern "C" void kernel_launch(void* const* d_in, const int* in_sizes, int n_in,
                              void* d_out, int out_size, void* d_ws, size_t ws_size,
                              hipStream_t stream) {
  (void)in_sizes; (void)n_in; (void)out_size; (void)ws_size;
  const int B = 64, S = 1024, I = 128, HID = 1024, O = 128;
  const int M = B * S;                // 65536 GEMM rows

  const float* x    = (const float*)d_in[0];
  const float* h0   = (const float*)d_in[1];
  const float* Wxz[2] = {(const float*)d_in[2],  (const float*)d_in[3]};
  const float* bxz[2] = {(const float*)d_in[4],  (const float*)d_in[5]};
  const float* Whz[2] = {(const float*)d_in[6],  (const float*)d_in[7]};
  const float* Wxr[2] = {(const float*)d_in[8],  (const float*)d_in[9]};
  const float* bxr[2] = {(const float*)d_in[10], (const float*)d_in[11]};
  const float* Whr[2] = {(const float*)d_in[12], (const float*)d_in[13]};
  const float* Wxg[2] = {(const float*)d_in[14], (const float*)d_in[15]};
  const float* bxg[2] = {(const float*)d_in[16], (const float*)d_in[17]};
  const float* Whg[2] = {(const float*)d_in[18], (const float*)d_in[19]};
  const float* Wout = (const float*)d_in[20];
  const float* bout = (const float*)d_in[21];

  // ---- workspace carve-up (256B aligned chunks) ----
  char* ws = (char*)d_ws;
  size_t off = 0;
  auto carve = [&](size_t bytes) -> char* {
    char* p = ws + off;
    off += (bytes + 255) & ~(size_t)255;
    return p;
  };
  h16* gZ   = (h16*)carve((size_t)M * HID * 2);   // x-side pre-activations (time-major)
  h16* gR   = (h16*)carve((size_t)M * HID * 2);
  h16* gG   = (h16*)carve((size_t)M * HID * 2);
  h16* buf0 = (h16*)carve((size_t)M * I * 2);     // x f16, time-major rows
  h16* buf1 = (h16*)carve((size_t)M * HID * 2);   // layer-0 output (f16, time-major)
  h16* buf2 = (h16*)carve((size_t)M * HID * 2);   // layer-1 output (f16, time-major)
  h16* wxzF[2] = {(h16*)carve(HID * I * 2), (h16*)carve((size_t)HID * HID * 2)};
  h16* wxrF[2] = {(h16*)carve(HID * I * 2), (h16*)carve((size_t)HID * HID * 2)};
  h16* wxgF[2] = {(h16*)carve(HID * I * 2), (h16*)carve((size_t)HID * HID * 2)};
  h16* whzF[2] = {(h16*)carve((size_t)HID * HID * 2), (h16*)carve((size_t)HID * HID * 2)};
  h16* whrF[2] = {(h16*)carve((size_t)HID * HID * 2), (h16*)carve((size_t)HID * HID * 2)};
  h16* whgF[2] = {(h16*)carve((size_t)HID * HID * 2), (h16*)carve((size_t)HID * HID * 2)};
  h16* woutF   = (h16*)carve((size_t)O * HID * 2);
  float* hF  = (float*)carve((size_t)B * HID * 4);
  float* zF  = (float*)carve((size_t)B * HID * 4);
  h16*   hH  = (h16*)carve((size_t)B * HID * 2);
  h16*   rhH = (h16*)carve((size_t)B * HID * 2);

  auto conv = [&](const float* s, h16* d, int n) {
    cvt_f16<<<(n + 255) / 256, 256, 0, stream>>>(s, d, n);
  };

  // ---- weight conversion to f16 ----
  conv(Wxz[0], wxzF[0], HID * I);  conv(Wxz[1], wxzF[1], HID * HID);
  conv(Wxr[0], wxrF[0], HID * I);  conv(Wxr[1], wxrF[1], HID * HID);
  conv(Wxg[0], wxgF[0], HID * I);  conv(Wxg[1], wxgF[1], HID * HID);
  conv(Whz[0], whzF[0], HID * HID); conv(Whz[1], whzF[1], HID * HID);
  conv(Whr[0], whrF[0], HID * HID); conv(Whr[1], whrF[1], HID * HID);
  conv(Whg[0], whgF[0], HID * HID); conv(Whg[1], whgF[1], HID * HID);
  conv(Wout, woutF, O * HID);

  x_cvt_tm<<<(M * I) / 256, 256, 0, stream>>>(x, buf0);

  float* out_y = (float*)d_out;                 // (B,S,O)
  float* out_h = out_y + (size_t)M * O;         // (B,L,HID)

  for (int l = 0; l < 2; ++l) {
    const h16* curIn = (l == 0) ? buf0 : buf1;
    h16* curOut      = (l == 0) ? buf1 : buf2;
    int K = (l == 0) ? I : HID;

    // x-side gate projections: (65536 x K) x (1024 x K)^T -> f16 + bias
    int waves  = (M >> 5) * (HID >> 6);         // (M/32)*(N/64)
    int blocks = (waves * 32) / 256;
    gemm_bias<true, false><<<blocks, 256, 0, stream>>>(curIn, wxzF[l], bxz[l], gZ, HID, K);
    gemm_bias<true, false><<<blocks, 256, 0, stream>>>(curIn, wxrF[l], bxr[l], gR, HID, K);
    gemm_bias<true, false><<<blocks, 256, 0, stream>>>(curIn, wxgF[l], bxg[l], gG, HID, K);

    h_init<<<(B * HID) / 256, 256, 0, stream>>>(h0, hF, hH, l);

    dim3 sgrid(64, 4);                          // 64 col tiles x 4 row tiles
    for (int t = 0; t < S; ++t) {
      gru_step_zr<<<sgrid, 128, 0, stream>>>(hH, whzF[l], whrF[l], gZ, gR,
                                             hF, zF, rhH, t);
      gru_step_g<<<sgrid, 128, 0, stream>>>(rhH, whgF[l], gG, zF,
                                            hF, hH, curOut, t);
    }
    h_last<<<(B * HID) / 256, 256, 0, stream>>>(hF, out_h, l);
  }

  // Output projection: (65536 x 1024) x (128 x 1024)^T -> f32 d_out + bias,
  // rows permuted back from time-major to (b*S+t).
  {
    int waves  = (M >> 5) * (O >> 6);
    int blocks = (waves * 32) / 256;
    gemm_bias<false, true><<<blocks, 256, 0, stream>>>(buf2, woutF, bout, out_y, O, HID);
  }
}
